// MoELayerParallel_70317204570821
// MI455X (gfx1250) — matrix operand order repaired and checked by reading, not compiled, not run on hardware
//
#include <hip/hip_runtime.h>
#include <hip/hip_bf16.h>

#define N_EMBD  1024
#define HIDDEN  4096
#define EXPERTS 8
#define TOKENS  16384
#define TILE_M  32
#define MAX_TILES (TOKENS / TILE_M)   // 512
#define XSTR 1032                     // x tile LDS row stride (bf16 elems), padded
#define HSTR 136                      // h tile LDS row stride (bf16 elems), padded
#define GATE_BLOCKS (TOKENS / 8)      // 2048
#define PKE ((size_t)HIDDEN * N_EMBD) // packed bf16 elems per expert (4M)

// dynamic LDS partition (bytes)
#define XS_BYTES  (TILE_M * XSTR * 2)              // 66048
#define HS_BYTES  (TILE_M * HSTR * 2)              // 8704
#define TOK_OFF   (XS_BYTES + HS_BYTES)            // 74752 (16-aligned)
#define GV_OFF    (TOK_OFF + TILE_M * 4)           // 74880
#define SMEM_BYTES (GV_OFF + TILE_M * 4)           // 75008

typedef __attribute__((ext_vector_type(16))) __bf16 v16bf;
typedef __attribute__((ext_vector_type(8)))  float  v8f;

union V16 { v16bf v; unsigned short s[16]; uint4 q[2]; };

__device__ __forceinline__ unsigned short f2bf(float f) {
  unsigned u = __builtin_bit_cast(unsigned, f);
  u += 0x7FFFu + ((u >> 16) & 1u);          // round-to-nearest-even
  return (unsigned short)(u >> 16);
}

// A-matrix (16x32 bf16) operand: two contiguous 8-element groups per lane.
__device__ __forceinline__ v16bf loadA_lds(const unsigned short* row, int off) {
  V16 r;
  r.q[0] = *(const uint4*)(row + off);       // K = 16*half + 0..7   (+d0)
  r.q[1] = *(const uint4*)(row + off + 8);   // K = 16*half + 8..15
  return r.v;
}

// B operand from packed bf16 workspace: lane's 16 values are 32 contiguous bytes.
__device__ __forceinline__ v16bf loadB_pk(const unsigned short* p) {
  V16 r;
  r.q[0] = ((const uint4*)p)[0];
  r.q[1] = ((const uint4*)p)[1];
  return r.v;
}

// Fallback: B operand from f32 weights with on-the-fly conversion.
__device__ __forceinline__ v16bf loadB_gl(const float* p) {
  V16 r;
#pragma unroll
  for (int i = 0; i < 4; ++i) {
    float4 f = ((const float4*)p)[i];
    r.s[i*4+0] = f2bf(f.x); r.s[i*4+1] = f2bf(f.y);
    r.s[i*4+2] = f2bf(f.z); r.s[i*4+3] = f2bf(f.w);
  }
  return r.v;
}

// ---------------- zero output + routing counters ----------------
__global__ void zero_kernel(float* __restrict__ out, size_t n, int* __restrict__ counts) {
  size_t i = (size_t)blockIdx.x * blockDim.x + threadIdx.x;
  size_t stride = (size_t)gridDim.x * blockDim.x;
  for (; i < n; i += stride) out[i] = 0.0f;
  if (blockIdx.x == 0 && threadIdx.x < EXPERTS) counts[threadIdx.x] = 0;
}

// ---------------- weight f32 -> bf16 swizzled pre-conversion ----------------
// W: [E][rows][cols] f32  ->  out: [E][rows/16][cols/32][lane32][16] bf16
// lane = half*16 + nn maps to matrix row rt*16+nn, k-window cc*32 + half*16 + 0..15,
// the per-lane v16bf B operand in the same contiguous k order (16*half + s) that the A loads use.
__global__ __launch_bounds__(256) void convert_w_kernel(
    const float* __restrict__ W, unsigned short* __restrict__ out,
    int rows, int cols) {
  size_t idx = (size_t)blockIdx.x * 256 + threadIdx.x;   // one per (e, rt, cc, lane)
  size_t total = (size_t)EXPERTS * (rows / 16) * (cols / 32) * 32;
  if (idx >= total) return;
  int lane = (int)(idx & 31);
  size_t rest = idx >> 5;
  int cc = (int)(rest % (size_t)(cols / 32));
  rest /= (size_t)(cols / 32);
  int rt = (int)(rest % (size_t)(rows / 16));
  int e  = (int)(rest / (size_t)(rows / 16));
  int half = lane >> 4, nn = lane & 15;

  const float* src = W + ((size_t)e * rows + (size_t)rt * 16 + nn) * cols + cc * 32 + half * 16;
  V16 r;
#pragma unroll
  for (int i = 0; i < 4; ++i) {
    float4 f = ((const float4*)src)[i];
    r.s[i*4+0] = f2bf(f.x); r.s[i*4+1] = f2bf(f.y);
    r.s[i*4+2] = f2bf(f.z); r.s[i*4+3] = f2bf(f.w);
  }
  uint4* dst = (uint4*)(out + idx * 16);
  dst[0] = r.q[0];
  dst[1] = r.q[1];
}

// ---------------- gate: logits, top-2 routing, lb partials ----------------
__global__ __launch_bounds__(256) void gate_kernel(
    const float* __restrict__ x, const float* __restrict__ gw,
    const float* __restrict__ nw, const float* __restrict__ noise,
    float* __restrict__ gate_out, int* __restrict__ lists,
    int* __restrict__ counts, float* __restrict__ partials) {
  __shared__ float probs_lds[8][EXPERTS];
  const int wave = threadIdx.x >> 5;
  const int lane = threadIdx.x & 31;
  const int t = blockIdx.x * 8 + wave;

  float acc[EXPERTS];
#pragma unroll
  for (int e = 0; e < EXPERTS; ++e) acc[e] = 0.0f;
  for (int d = lane; d < N_EMBD; d += 32) {
    float xv = x[(size_t)t * N_EMBD + d];
#pragma unroll
    for (int e = 0; e < EXPERTS; ++e) acc[e] += xv * gw[e * N_EMBD + d];
  }
#pragma unroll
  for (int e = 0; e < EXPERTS; ++e) {
    float v = acc[e];
#pragma unroll
    for (int o = 16; o > 0; o >>= 1) v += __shfl_xor(v, o, 32);
    acc[e] = v;
  }

  if (lane == 0) {
    float mx = acc[0];
#pragma unroll
    for (int e = 1; e < EXPERTS; ++e) mx = fmaxf(mx, acc[e]);
    float p[EXPERTS], s = 0.0f;
#pragma unroll
    for (int e = 0; e < EXPERTS; ++e) { p[e] = __expf(acc[e] - mx); s += p[e]; }
    float inv = 1.0f / s;
#pragma unroll
    for (int e = 0; e < EXPERTS; ++e) probs_lds[wave][e] = p[e] * inv;

    float ln[EXPERTS];
#pragma unroll
    for (int e = 0; e < EXPERTS; ++e)
      ln[e] = acc[e] + noise[(size_t)t * EXPERTS + e] * nw[e];

    int i1 = 0;
#pragma unroll
    for (int e = 1; e < EXPERTS; ++e) if (ln[e] > ln[i1]) i1 = e;
    int i2 = (i1 == 0) ? 1 : 0;
#pragma unroll
    for (int e = 0; e < EXPERTS; ++e) if (e != i1 && ln[e] > ln[i2]) i2 = e;

    float e2 = __expf(ln[i2] - ln[i1]);
    float g1 = 1.0f / (1.0f + e2);
    float g2 = e2 / (1.0f + e2);
#pragma unroll
    for (int e = 0; e < EXPERTS; ++e)
      gate_out[(size_t)t * EXPERTS + e] = (e == i1) ? g1 : ((e == i2) ? g2 : 0.0f);

    int p1 = atomicAdd(&counts[i1], 1); lists[i1 * TOKENS + p1] = t;
    int p2 = atomicAdd(&counts[i2], 1); lists[i2 * TOKENS + p2] = t;
  }
  __syncthreads();
  if (threadIdx.x < EXPERTS) {
    float s = 0.0f;
#pragma unroll
    for (int w = 0; w < 8; ++w) s += probs_lds[w][threadIdx.x];  // fixed order
    partials[blockIdx.x * EXPERTS + threadIdx.x] = s;
  }
}

// ---------------- lb loss: fixed-order reduction ----------------
__global__ void lb_kernel(const float* __restrict__ partials, float* __restrict__ out_scalar) {
  __shared__ float sq[EXPERTS];
  if (threadIdx.x < EXPERTS) {
    float s = 0.0f;
    for (int b = 0; b < GATE_BLOCKS; ++b) s += partials[b * EXPERTS + threadIdx.x];
    float m = s * (1.0f / (float)TOKENS) - (1.0f / (float)EXPERTS);
    sq[threadIdx.x] = m * m;
  }
  __syncthreads();
  if (threadIdx.x == 0) {
    float s = 0.0f;
#pragma unroll
    for (int e = 0; e < EXPERTS; ++e) s += sq[e];
    *out_scalar = s * (0.01f / (float)EXPERTS);
  }
}

// ---------------- fused SwiGLU expert FFN, WMMA bf16, M=32 token tiles ----------------
// 512 threads = 16 waves. Wave (m, ws): m = M-half (16 tokens), ws = 128-col slice.
// Step A: each wave computes one 16x16 h tile (its jt slice, its M-half).
// Step B: each wave accumulates y[m][ws*128 .. +127] (yacc = 8 x v8f = 64 VGPRs).
template <bool PK>
__global__ __launch_bounds__(512) void moe_ffn_kernel(
    const float* __restrict__ x,
    const float* __restrict__ W1, const float* __restrict__ b1,
    const float* __restrict__ W2, const float* __restrict__ b2,
    const float* __restrict__ Wp, const float* __restrict__ bp,
    const unsigned short* __restrict__ W1p, const unsigned short* __restrict__ W2p,
    const unsigned short* __restrict__ Wpp,
    const float* __restrict__ gatew, const int* __restrict__ lists,
    const int* __restrict__ counts, float* __restrict__ out) {
  extern __shared__ __align__(16) char smem[];
  unsigned short* xs = (unsigned short*)smem;                 // [TILE_M][XSTR] bf16
  unsigned short* hs = (unsigned short*)(smem + XS_BYTES);    // [TILE_M][HSTR] bf16
  int*   toks  = (int*)(smem + TOK_OFF);                      // [TILE_M]
  float* gvals = (float*)(smem + GV_OFF);                     // [TILE_M]

  const int e    = blockIdx.x >> 9;              // / MAX_TILES
  const int tile = blockIdx.x & (MAX_TILES - 1);
  const int cnt  = counts[e];
  const int tb   = tile * TILE_M;
  if (tb >= cnt) return;                          // block-uniform

  const int tid  = threadIdx.x;
  const int w2   = tid >> 5;                      // wave 0..15
  const int m    = w2 >> 3;                       // M-half: 0 or 1
  const int ws   = w2 & 7;                        // 128-col / 16-hidden slice
  const int lane = tid & 31;
  const int half = lane >> 4;
  const int nn   = lane & 15;

  if (tid < TILE_M) {
    int slot = tb + tid;
    int tk = (slot < cnt) ? lists[e * TOKENS + slot] : -1;
    toks[tid]  = tk;
    gvals[tid] = (tk >= 0) ? gatew[(size_t)tk * EXPERTS + e] : 0.0f;
  }
  __syncthreads();

  // stage x tile (f32 -> bf16) into LDS: 32 rows x 1024
  for (int idx = tid; idx < TILE_M * (N_EMBD / 4); idx += 512) {
    int row = idx >> 8;
    int c4  = idx & 255;
    int tk  = toks[row];
    float4 v = make_float4(0.f, 0.f, 0.f, 0.f);
    if (tk >= 0) v = ((const float4*)(x + (size_t)tk * N_EMBD))[c4];
    ushort4 pk;
    pk.x = f2bf(v.x); pk.y = f2bf(v.y); pk.z = f2bf(v.z); pk.w = f2bf(v.w);
    *(ushort4*)&xs[row * XSTR + c4 * 4] = pk;
  }
  __syncthreads();

  const float* W1e = W1 + (size_t)e * HIDDEN * N_EMBD;
  const float* W2e = W2 + (size_t)e * HIDDEN * N_EMBD;
  const float* Wpe = Wp + (size_t)e * N_EMBD * HIDDEN;
  const unsigned short* W1pe = W1p + (size_t)e * PKE;
  const unsigned short* W2pe = W2p + (size_t)e * PKE;
  const unsigned short* Wppe = Wpp + (size_t)e * PKE;

  v8f yacc[8];
#pragma unroll
  for (int i = 0; i < 8; ++i) yacc[i] = (v8f)0.0f;

  const unsigned short* xrow = xs + (m * 16 + nn) * XSTR;  // A row for this lane
  const unsigned short* hrow = hs + (m * 16 + nn) * HSTR;

  for (int c = 0; c < HIDDEN / 128; ++c) {
    const int j0 = c * 128;
    const int jw = j0 + ws * 16;                  // this wave's hidden columns
    const int jt = c * 8 + ws;                    // hidden 16-tile index

    // ---- step A: a1 = x@W1^T, a2 = x@W2^T  (K = 1024), one 16x16 tile/wave ----
    v8f a1 = (v8f)0.0f, a2 = (v8f)0.0f;
    const float* w1row = W1e + (size_t)(jw + nn) * N_EMBD + half * 16;
    const float* w2row = W2e + (size_t)(jw + nn) * N_EMBD + half * 16;
    const unsigned short* p1 = W1pe + ((size_t)jt * (N_EMBD / 32) * 32 + lane) * 16;
    const unsigned short* p2 = W2pe + ((size_t)jt * (N_EMBD / 32) * 32 + lane) * 16;
#pragma unroll 4
    for (int ks = 0; ks < N_EMBD / 32; ++ks) {
      const int d0 = ks * 32;
      v16bf A = loadA_lds(xrow, d0 + half * 16);
      v16bf B1, B2;
      if constexpr (PK) {
        B1 = loadB_pk(p1 + (size_t)ks * 512);     // 32 contiguous bytes/lane
        B2 = loadB_pk(p2 + (size_t)ks * 512);
      } else {
        B1 = loadB_gl(w1row + d0);
        B2 = loadB_gl(w2row + d0);
      }
      a1 = __builtin_amdgcn_wmma_f32_16x16x32_bf16(false, A, false, B1, (short)0, a1, false, false);
      a2 = __builtin_amdgcn_wmma_f32_16x16x32_bf16(false, A, false, B2, (short)0, a2, false, false);
    }
    const float bb1 = b1[e * HIDDEN + jw + nn];
    const float bb2 = b2[e * HIDDEN + jw + nn];
    float hv[8];
#pragma unroll
    for (int r = 0; r < 8; ++r) {
      float z1 = a1[r] + bb1;
      float z2 = a2[r] + bb2;
      hv[r] = z1 * (z2 / (1.0f + __expf(-z2)));   // z1 * silu(z2)
    }
    __syncthreads();                              // prior step-B readers done
#pragma unroll
    for (int r = 0; r < 8; ++r)
      hs[(m * 16 + r + 8 * half) * HSTR + ws * 16 + nn] = f2bf(hv[r]);
    __syncthreads();

    // ---- step B: y += h_chunk @ Wp_chunk^T ----
#pragma unroll
    for (int kb = 0; kb < 4; ++kb) {
      const int d0b = kb * 32;
      const int kcc = c * 4 + kb;                 // k-chunk index into HIDDEN
      v16bf Ah = loadA_lds(hrow, d0b + half * 16);
#pragma unroll
      for (int nt = 0; nt < 8; ++nt) {
        const int ncol = ws * 128 + nt * 16 + nn;
        v16bf Bp;
        if constexpr (PK) {
          const int ct = ws * 8 + nt;             // out-col 16-tile index
          Bp = loadB_pk(Wppe + (((size_t)ct * (HIDDEN / 32) + kcc) * 32 + lane) * 16);
        } else {
          Bp = loadB_gl(Wpe + (size_t)ncol * HIDDEN + j0 + d0b + half * 16);
        }
        yacc[nt] = __builtin_amdgcn_wmma_f32_16x16x32_bf16(false, Ah, false, Bp, (short)0, yacc[nt], false, false);
      }
    }
  }

  // ---- epilogue: out[token] += g * (y + bp) ----
#pragma unroll
  for (int nt = 0; nt < 8; ++nt) {
    const int ncol = ws * 128 + nt * 16 + nn;
    const float bpn = bp[e * N_EMBD + ncol];
#pragma unroll
    for (int r = 0; r < 8; ++r) {
      const int row = m * 16 + r + 8 * half;      // C/D layout: M = r + 8*(lane/16)
      if (tb + row < cnt) {
        float val = gvals[row] * (yacc[nt][r] + bpn);
        unsafeAtomicAdd(out + (size_t)toks[row] * N_EMBD + ncol, val);
      }
    }
  }
}

extern "C" void kernel_launch(void* const* d_in, const int* in_sizes, int n_in,
                              void* d_out, int out_size, void* d_ws, size_t ws_size,
                              hipStream_t stream) {
  const float* x     = (const float*)d_in[0];
  const float* gw    = (const float*)d_in[1];
  const float* nwgt  = (const float*)d_in[2];
  const float* noise = (const float*)d_in[3];
  const float* W1    = (const float*)d_in[4];
  const float* b1    = (const float*)d_in[5];
  const float* W2    = (const float*)d_in[6];
  const float* b2    = (const float*)d_in[7];
  const float* Wp    = (const float*)d_in[8];
  const float* bp    = (const float*)d_in[9];
  float* out = (float*)d_out;

  char* ws = (char*)d_ws;
  // misc routing state in first 2 MB
  int*   counts   = (int*)ws;                                     // 8 ints
  int*   lists    = (int*)(ws + 32);                              // 8*T ints
  float* gatews   = (float*)(ws + 32 + (size_t)EXPERTS*TOKENS*4); // T*8 floats
  float* partials = (float*)(ws + 32 + (size_t)EXPERTS*TOKENS*4
                                  + (size_t)TOKENS*EXPERTS*4);    // GATE_BLOCKS*8
  // packed bf16 weights at +2 MB
  const size_t WBASE = 2u * 1024 * 1024;
  const size_t WBYTES = PKE * EXPERTS * 2;                        // 64 MB each
  unsigned short* w1bf = (unsigned short*)(ws + WBASE);
  unsigned short* w2bf = (unsigned short*)(ws + WBASE + WBYTES);
  unsigned short* wpbf = (unsigned short*)(ws + WBASE + 2 * WBYTES);
  const bool use_pk = ws_size >= WBASE + 3 * WBYTES;

  // 1) zero output accumulator + routing counters
  zero_kernel<<<2048, 256, 0, stream>>>(out, (size_t)out_size, counts);

  // 2) one-shot weight f32->bf16 swizzled conversion (~600 MB of traffic, ~26 us)
  if (use_pk) {
    const int cblocks = (int)(((size_t)EXPERTS * (HIDDEN/16) * (N_EMBD/32) * 32 + 255) / 256);
    convert_w_kernel<<<cblocks, 256, 0, stream>>>(W1, w1bf, HIDDEN, N_EMBD);
    convert_w_kernel<<<cblocks, 256, 0, stream>>>(W2, w2bf, HIDDEN, N_EMBD);
    const int pblocks = (int)(((size_t)EXPERTS * (N_EMBD/16) * (HIDDEN/32) * 32 + 255) / 256);
    convert_w_kernel<<<pblocks, 256, 0, stream>>>(Wp, wpbf, N_EMBD, HIDDEN);
  }

  // 3) gate: logits, top-2 routing, lb partial sums
  gate_kernel<<<GATE_BLOCKS, 256, 0, stream>>>(x, gw, nwgt, noise,
                                               gatews, lists, counts, partials);
  // 4) load-balance loss scalar (deterministic fixed-order reduction)
  lb_kernel<<<1, 64, 0, stream>>>(partials, out + (size_t)TOKENS * N_EMBD);

  // 5) routed fused SwiGLU FFN + weighted scatter-add (M=32 tiles, 75 KB dynamic LDS)
  if (use_pk) {
    moe_ffn_kernel<true><<<EXPERTS * MAX_TILES, 512, SMEM_BYTES, stream>>>(
        x, W1, b1, W2, b2, Wp, bp, w1bf, w2bf, wpbf, gatews, lists, counts, out);
  } else {
    moe_ffn_kernel<false><<<EXPERTS * MAX_TILES, 512, SMEM_BYTES, stream>>>(
        x, W1, b1, W2, b2, Wp, bp, w1bf, w2bf, wpbf, gatews, lists, counts, out);
  }
}
